// bottleneck_IR_Deform_27367531610883
// MI455X (gfx1250) — matrix operand order repaired and checked by reading, not compiled, not run on hardware
//
#include <hip/hip_runtime.h>
#include <hip/hip_bf16.h>

typedef __attribute__((ext_vector_type(16))) __bf16 v16bf;
typedef __attribute__((ext_vector_type(8)))  float  v8f;

#define CCH   128
#define HH    128
#define WW    128
#define HW    (HH*WW)
#define KTAP  9
#define CK    (CCH*KTAP)      // 1152
#define TILE_N 128            // deform-GEMM N tile
#define ON     256            // offmod-GEMM N tile

union AFrag { v16bf v; uint4 q[2]; };

// ---------------- elementwise bn1 ----------------
__global__ __launch_bounds__(256) void k_bn(const float* __restrict__ x,
                                            const float* __restrict__ g,
                                            const float* __restrict__ b,
                                            const float* __restrict__ m,
                                            const float* __restrict__ v,
                                            float* __restrict__ out) {
    size_t i = (size_t)blockIdx.x * 256 + threadIdx.x;
    int c = (int)((i >> 14) & 127);
    out[i] = (x[i] - m[c]) * (g[c] * rsqrtf(v[c] + 1e-5f)) + b[c];
}

// ---- reorder GEMM weights [O=128][C][9] f32 -> [O][9][C] bf16 ----
__global__ __launch_bounds__(256) void k_reorder(const float* __restrict__ w,
                                                 __bf16* __restrict__ wbf) {
    int i = blockIdx.x * 256 + threadIdx.x;
    if (i >= CCH * CK) return;
    int o = i / CK, rem = i % CK;
    int c = rem / KTAP, k = rem % KTAP;
    wbf[(o * KTAP + k) * CCH + c] = (__bf16)w[i];
}

// ---- merge+reorder offset/mod weights -> [32(pad)][9][128] bf16, rows 27..31 = 0 ----
__global__ __launch_bounds__(256) void k_prep_womb(const float* __restrict__ offw,
                                                   const float* __restrict__ modw,
                                                   __bf16* __restrict__ womb) {
    int i = blockIdx.x * 256 + threadIdx.x;          // 32*9*128
    if (i >= 32 * KTAP * CCH) return;
    int o = i / (KTAP * CCH), rem = i % (KTAP * CCH);
    int k = rem / CCH, c = rem % CCH;
    float v = 0.f;
    if (o < 18)      v = offw[(o * CCH + c) * KTAP + k];
    else if (o < 27) v = modw[((o - 18) * CCH + c) * KTAP + k];
    womb[i] = (__bf16)v;
}

// ---------------- offset/mod conv as WMMA implicit GEMM ----------------
// M = 32 (27 live rows), K = 1152, workgroup tile = 32M x 256N
__global__ __launch_bounds__(256) void k_offmod_wmma(
        const float* __restrict__ rin,      // [B][128][HW]
        const __bf16* __restrict__ womb,    // [32][9][128]
        const float* __restrict__ offb,     // [18]
        const float* __restrict__ modb,     // [9]
        float* __restrict__ om) {           // [B][27][HW]
    __shared__ unsigned int sB[ON * 20];    // [n][k/2], 80B pitch -> 20KB

    const int tid   = threadIdx.x;
    const int t     = blockIdx.x;           // B*HW/ON = 256 blocks
    const int b     = t >> 6;
    const int base  = (t & 63) * ON;
    const int nprod = tid;                  // one n column per thread
    const int pos   = base + nprod;
    const int h = pos >> 7, w = pos & 127;
    const float* rb = rin + (size_t)b * CCH * HW;

    v8f acc[4] = {};
    const int wv   = tid >> 5, lane = tid & 31;
    const int mrow = (wv >> 2) * 16;        // 0 or 16
    const int nsub = (wv & 3) * 64;
    const int half = (lane >> 4) << 3;

    for (int k = 0; k < KTAP; ++k) {
        int y = h + (k / 3) - 1;
        int x = w + (k % 3) - 1;
        bool valid = ((unsigned)y < (unsigned)HH) && ((unsigned)x < (unsigned)WW);
        int  sidx  = valid ? (y * WW + x) : 0;
        float vmul = valid ? 1.f : 0.f;
        for (int cb = 0; cb < 4; ++cb) {
            int cbase = cb * 32;
            __syncthreads();
#pragma unroll
            for (int p = 0; p < 16; ++p) {
                int c = cbase + p * 2;
                float v0 = rb[(size_t)c * HW + sidx] * vmul;
                float v1 = rb[(size_t)(c + 1) * HW + sidx] * vmul;
                unsigned short u0 = __builtin_bit_cast(unsigned short, (__bf16)v0);
                unsigned short u1 = __builtin_bit_cast(unsigned short, (__bf16)v1);
                sB[nprod * 20 + p] = ((unsigned)u1 << 16) | (unsigned)u0;
            }
            __syncthreads();
            int kb = k * CCH + cbase;
            int m  = mrow + (lane & 15);
            int abase = m * CK + kb + ((lane >> 4) << 4);
            AFrag a;
            a.q[0] = *reinterpret_cast<const uint4*>(womb + abase);
            a.q[1] = *reinterpret_cast<const uint4*>(womb + abase + 8);
#pragma unroll
            for (int j = 0; j < 4; ++j) {
                int n = nsub + (lane & 15) + j * 16;
                AFrag bb;
                const uint4* bp = reinterpret_cast<const uint4*>(&sB[n * 20 + half]);
                bb.q[0] = bp[0];
                bb.q[1] = bp[1];
                acc[j] = __builtin_amdgcn_wmma_f32_16x16x32_bf16(
                    false, a.v, false, bb.v, (short)0, acc[j], false, false);
            }
        }
    }

    const int ncol0 = lane & 15;
    float* ob = om + (size_t)b * 27 * HW;
#pragma unroll
    for (int r = 0; r < 8; ++r) {
        int M = mrow + ((lane >> 4) << 3) + r;
        if (M < 27) {
            float bias = (M < 18) ? offb[M] : modb[M - 18];
#pragma unroll
            for (int j = 0; j < 4; ++j) {
                int p2 = base + nsub + ncol0 + j * 16;
                ob[(size_t)M * HW + p2] = acc[j][r] + bias;
            }
        }
    }
}

// ---------------- fused deformable-conv implicit GEMM (WMMA bf16) ----------------
// workgroup tile = 128M x 128N, K-chunks = 9 taps x 4 channel-blocks
__global__ __launch_bounds__(256) void k_deform_gemm(
        const float* __restrict__ rin,   // [B][128][HW] sampled source
        const float* __restrict__ om,    // [B][27][HW]  offsets+mask logits
        const __bf16* __restrict__ wbf,  // [128][9][128] bf16
        float* __restrict__ out,         // [B][128][HW]
        const float* __restrict__ xres,  // shortcut (mode 1) or null
        const float* __restrict__ p0,    // alpha / bn g
        const float* __restrict__ p1,    // bn b
        const float* __restrict__ p2,    // bn m
        const float* __restrict__ p3,    // bn v
        int mode) {                      // 0 = prelu, 1 = bn + residual
    __shared__ short sY0[KTAP * TILE_N], sY1[KTAP * TILE_N];
    __shared__ short sX0[KTAP * TILE_N], sX1[KTAP * TILE_N];
    __shared__ float sWq[KTAP * TILE_N * 4];
    __shared__ unsigned int sB[TILE_N * 20];     // [n][k/2] pitch 20 dwords

    const int tid  = threadIdx.x;
    const int t    = blockIdx.x;
    const int b    = t >> 7;                     // 128 tiles per batch
    const int base = (t & 127) * TILE_N;

    // ---- bilinear metadata, shared across all 128 channels ----
    const float* omb = om + (size_t)b * 27 * HW;
    for (int i = tid; i < KTAP * TILE_N; i += 256) {
        int k = i >> 7, n = i & 127;
        int pos = base + n;
        int h = pos >> 7, w = pos & 127;
        float oy = omb[(size_t)(k * 2 + 0) * HW + pos];
        float ox = omb[(size_t)(k * 2 + 1) * HW + pos];
        float mk = omb[(size_t)(18 + k) * HW + pos];
        float mv = 2.0f / (1.0f + __expf(-mk));
        float py = (float)(h - 1 + (k / 3)) + oy;
        float px = (float)(w - 1 + (k % 3)) + ox;
        float fy = floorf(py), fx = floorf(px);
        float ay = py - fy, ax = px - fx;
        int y0 = (int)fy, x0 = (int)fx;
        int y1 = y0 + 1, x1 = x0 + 1;
        float vy0 = (y0 >= 0 && y0 < HH) ? 1.f : 0.f;
        float vy1 = (y1 >= 0 && y1 < HH) ? 1.f : 0.f;
        float vx0 = (x0 >= 0 && x0 < WW) ? 1.f : 0.f;
        float vx1 = (x1 >= 0 && x1 < WW) ? 1.f : 0.f;
        sWq[i * 4 + 0] = (1.f - ay) * (1.f - ax) * vy0 * vx0 * mv;
        sWq[i * 4 + 1] = (1.f - ay) * ax * vy0 * vx1 * mv;
        sWq[i * 4 + 2] = ay * (1.f - ax) * vy1 * vx0 * mv;
        sWq[i * 4 + 3] = ay * ax * vy1 * vx1 * mv;
        sY0[i] = (short)min(max(y0, 0), HH - 1);
        sY1[i] = (short)min(max(y1, 0), HH - 1);
        sX0[i] = (short)min(max(x0, 0), WW - 1);
        sX1[i] = (short)min(max(x1, 0), WW - 1);
    }

    v8f acc[8] = {};
    const int wv    = tid >> 5;
    const int lane  = tid & 31;
    const int nprod = tid & 127;      // producer spatial column
    const int pgrp  = tid >> 7;       // producer dword group 0..1
    const float* rb = rin + (size_t)b * CCH * HW;
    const int half  = (lane >> 4) << 3;

    for (int k = 0; k < KTAP; ++k) {
        int mi = k * TILE_N + nprod;
        short y0 = sY0[mi], y1 = sY1[mi], x0 = sX0[mi], x1 = sX1[mi];
        float w00 = sWq[mi * 4 + 0], w01 = sWq[mi * 4 + 1];
        float w10 = sWq[mi * 4 + 2], w11 = sWq[mi * 4 + 3];
        int i00 = (int)y0 * WW + x0, i01 = (int)y0 * WW + x1;
        int i10 = (int)y1 * WW + x0, i11 = (int)y1 * WW + x1;

        for (int cb = 0; cb < 4; ++cb) {
            int cbase = cb * 32;
            __syncthreads();
            // ---- produce 32(K) x 128(N) bf16 tile via bilinear gather ----
#pragma unroll
            for (int it = 0; it < 8; ++it) {
                int p = pgrp * 8 + it;                // dword index 0..15 (K pair)
                int c = cbase + p * 2;
                const float* pl0 = rb + (size_t)c * HW;
                const float* pl1 = pl0 + HW;
                float v0 = w00 * pl0[i00] + w01 * pl0[i01] + w10 * pl0[i10] + w11 * pl0[i11];
                float v1 = w00 * pl1[i00] + w01 * pl1[i01] + w10 * pl1[i10] + w11 * pl1[i11];
                unsigned short u0 = __builtin_bit_cast(unsigned short, (__bf16)v0);
                unsigned short u1 = __builtin_bit_cast(unsigned short, (__bf16)v1);
                sB[nprod * 20 + p] = ((unsigned)u1 << 16) | (unsigned)u0;
            }
            __syncthreads();

            // ---- consume: A once per chunk, 8x WMMA over N ----
            int kb = k * CCH + cbase;
            int m  = wv * 16 + (lane & 15);
            int abase = m * CK + kb + ((lane >> 4) << 4);
            AFrag a;
            a.q[0] = *reinterpret_cast<const uint4*>(wbf + abase);
            a.q[1] = *reinterpret_cast<const uint4*>(wbf + abase + 8);
#pragma unroll
            for (int j = 0; j < 8; ++j) {
                int n = (lane & 15) + j * 16;
                AFrag bb;
                const uint4* bp = reinterpret_cast<const uint4*>(&sB[n * 20 + half]);
                bb.q[0] = bp[0];
                bb.q[1] = bp[1];
                acc[j] = __builtin_amdgcn_wmma_f32_16x16x32_bf16(
                    false, a.v, false, bb.v, (short)0, acc[j], false, false);
            }
        }
    }

    // ---- epilogue ----
    const int mrow  = wv * 16 + ((lane >> 4) << 3);
    const int ncol0 = lane & 15;
#pragma unroll
    for (int r = 0; r < 8; ++r) {
        int M = mrow + r;
        float al = 0.f, sc = 0.f, bm = 0.f, bb2 = 0.f;
        if (mode == 0) { al = p0[M]; }
        else { sc = p0[M] * rsqrtf(p3[M] + 1e-5f); bm = p2[M]; bb2 = p1[M]; }
#pragma unroll
        for (int j = 0; j < 8; ++j) {
            int pos = base + ncol0 + j * 16;
            size_t oi = (size_t)(b * CCH + M) * HW + pos;
            float v = acc[j][r];
            if (mode == 0) v = v > 0.f ? v : al * v;
            else           v = (v - bm) * sc + bb2 + xres[oi];
            out[oi] = v;
        }
    }
}

extern "C" void kernel_launch(void* const* d_in, const int* in_sizes, int n_in,
                              void* d_out, int out_size, void* d_ws, size_t ws_size,
                              hipStream_t stream) {
    const float* x      = (const float*)d_in[0];
    const float* bn1_g  = (const float*)d_in[1];
    const float* bn1_b  = (const float*)d_in[2];
    const float* bn1_m  = (const float*)d_in[3];
    const float* bn1_v  = (const float*)d_in[4];
    const float* off1_w = (const float*)d_in[5];
    const float* off1_b = (const float*)d_in[6];
    const float* mod1_w = (const float*)d_in[7];
    const float* mod1_b = (const float*)d_in[8];
    const float* w1     = (const float*)d_in[9];
    const float* alpha  = (const float*)d_in[10];
    const float* off2_w = (const float*)d_in[11];
    const float* off2_b = (const float*)d_in[12];
    const float* mod2_w = (const float*)d_in[13];
    const float* mod2_b = (const float*)d_in[14];
    const float* w2     = (const float*)d_in[15];
    const float* bn2_g  = (const float*)d_in[16];
    const float* bn2_b  = (const float*)d_in[17];
    const float* bn2_m  = (const float*)d_in[18];
    const float* bn2_v  = (const float*)d_in[19];

    char* ws = (char*)d_ws;
    float*  r1    = (float*)(ws + 0);                   // 33.55 MB
    float*  r2    = (float*)(ws + 33554432);            // 33.55 MB
    float*  om1   = (float*)(ws + 67108864);            // 7.08 MB
    float*  om2   = (float*)(ws + 74186752);            // 7.08 MB
    __bf16* wbf1  = (__bf16*)(ws + 81264640);           // 288 KB
    __bf16* wbf2  = (__bf16*)(ws + 81559552);           // 288 KB
    __bf16* womb1 = (__bf16*)(ws + 81854464);           // 72 KB
    __bf16* womb2 = (__bf16*)(ws + 81928192);           // 72 KB

    const int NTOT = 4 * CCH * HW;                      // 8388608

    k_bn<<<NTOT / 256, 256, 0, stream>>>(x, bn1_g, bn1_b, bn1_m, bn1_v, r1);

    k_reorder<<<(CCH * CK + 255) / 256, 256, 0, stream>>>(w1, wbf1);
    k_reorder<<<(CCH * CK + 255) / 256, 256, 0, stream>>>(w2, wbf2);
    k_prep_womb<<<(32 * KTAP * CCH + 255) / 256, 256, 0, stream>>>(off1_w, mod1_w, womb1);
    k_prep_womb<<<(32 * KTAP * CCH + 255) / 256, 256, 0, stream>>>(off2_w, mod2_w, womb2);

    k_offmod_wmma<<<4 * (HW / ON), 256, 0, stream>>>(r1, womb1, off1_b, mod1_b, om1);
    k_deform_gemm<<<4 * (HW / TILE_N), 256, 0, stream>>>(
        r1, om1, wbf1, r2, nullptr, alpha, nullptr, nullptr, nullptr, 0);

    k_offmod_wmma<<<4 * (HW / ON), 256, 0, stream>>>(r2, womb2, off2_b, mod2_b, om2);
    k_deform_gemm<<<4 * (HW / TILE_N), 256, 0, stream>>>(
        r2, om2, wbf2, (float*)d_out, x, bn2_g, bn2_b, bn2_m, bn2_v, 1);
}
